// SeparateHiddenPRADAEncoder_369367188154
// MI455X (gfx1250) — matrix operand order repaired and checked by reading, not compiled, not run on hardware
//
#include <hip/hip_runtime.h>
#include <hip/hip_bf16.h>
#include <math.h>

// ---------------------------------------------------------------------------
// Types for CDNA5 WMMA
// ---------------------------------------------------------------------------
typedef __attribute__((ext_vector_type(16))) __bf16 v16bf;
typedef __attribute__((ext_vector_type(8)))  float  v8f;

// ---------------------------------------------------------------------------
// Small utility kernels
// ---------------------------------------------------------------------------
__global__ void fill_f32x4(float4* __restrict__ p, float v, long long n4) {
    long long i = (long long)blockIdx.x * blockDim.x + threadIdx.x;
    if (i < n4) p[i] = make_float4(v, v, v, v);
}

__global__ void deg_accum(const long long* __restrict__ col,
                          float* __restrict__ deg, int E) {
    int e = blockIdx.x * blockDim.x + threadIdx.x;
    if (e < E) unsafeAtomicAdd(&deg[(int)col[e]], 1.0f);
}

__global__ void deg_rsqrt(float* __restrict__ deg, int n) {
    int i = blockIdx.x * blockDim.x + threadIdx.x;
    if (i < n) deg[i] = rsqrtf(deg[i]);   // deg >= 1 always (self loop)
}

// ---------------------------------------------------------------------------
// Pack f32 weights (optionally two concatenated column blocks) into bf16
// WMMA B-fragment layout:
//   Wp[((nt*Kt + kt)*32 + lane)*16 + e] = W[k, j]   (bf16)
//   with k = kt*32 + (lane>>4)*16 + e,  j = nt*16 + (lane&15)
// Column j < split comes from W1 (row stride = split), else W2 (stride Nw-split).
// ---------------------------------------------------------------------------
__global__ void pack_w_bf16(const float* __restrict__ W1,
                            const float* __restrict__ W2,
                            int split, int K, int Nw,
                            __bf16* __restrict__ Wp) {
    int idx = blockIdx.x * blockDim.x + threadIdx.x;
    const int Kt = K >> 5;
    const int total = (Nw >> 4) * Kt * 32 * 16;
    if (idx >= total) return;
    int e  = idx & 15;
    int l  = (idx >> 4) & 31;
    int t  = idx >> 9;              // nt*Kt + kt
    int kt = t % Kt;
    int nt = t / Kt;
    int k  = kt * 32 + (l >> 4) * 16 + e;
    int j  = nt * 16 + (l & 15);
    float v = (j < split) ? W1[k * split + j]
                          : W2[k * (Nw - split) + (j - split)];
    Wp[idx] = (__bf16)v;
}

// ---------------------------------------------------------------------------
// Dense GEMM:  C[m, col0+n] = sum_k A[m,k] * W[k,n]    (bf16 WMMA, f32 acc)
//   A:  [M, K] f32 row-major, M = 16*MT*gridDim.x, K % 32 == 0
//   Wp: pre-packed bf16 B fragments (see pack_w_bf16), Nw % 16 == 0
// One wave computes MT stacked 16x16 tiles sharing each B fragment.
// Block = (32, 8) waves covering 8 n-tiles (Nw=128).
// ---------------------------------------------------------------------------
template <int MT>
__global__ void __launch_bounds__(256)
gemm_bf16_wmma(const float* __restrict__ A, int K,
               const __bf16* __restrict__ Wp, int Nw,
               float* __restrict__ C, int ldc, int col0) {
    const int lane = threadIdx.x;                 // 0..31 (wave32)
    const int half = lane >> 4;                   // 0 or 1
    const int m0   = blockIdx.x * (16 * MT);
    const int nt   = blockIdx.y * blockDim.y + threadIdx.y;
    if (nt * 16 >= Nw) return;

    const int Kt = K >> 5;
    const __bf16* __restrict__ wp =
        Wp + ((long long)(nt * Kt) * 32 + lane) * 16;
    const float* __restrict__ arow0 = A + (long long)(m0 + (lane & 15)) * K;

    v8f acc[MT];
#pragma unroll
    for (int mt = 0; mt < MT; ++mt) acc[mt] = (v8f){};

    for (int kt = 0; kt < Kt; ++kt) {
        const int k0 = kt << 5;
        // B fragment: one contiguous 32B-per-lane load (2 x b128)
        const v16bf b = *(const v16bf*)(wp + (long long)kt * 32 * 16);
#pragma unroll
        for (int mt = 0; mt < MT; ++mt) {
            const float* __restrict__ arow = arow0 + (long long)mt * 16 * K;
            // A fragment, 16x32 bf16, contiguous k order (matches packed B):
            //  lanes 0-15 : e0-15 -> K=k0+0..15
            //  lanes 16-31: e0-15 -> K=k0+16..31
            v16bf a{};
#pragma unroll
            for (int e = 0; e < 8; ++e) {
                a[e]     = (__bf16)arow[k0 + half * 16 + e];
                a[e + 8] = (__bf16)arow[k0 + half * 16 + 8 + e];
            }
            acc[mt] = __builtin_amdgcn_wmma_f32_16x16x32_bf16(
                /*neg_a=*/false, a, /*neg_b=*/false, b,
                /*c_mod=*/(short)0, acc[mt], /*reuse_a=*/false, /*reuse_b=*/false);
        }
    }

    // C/D layout: lanes 0-15 -> rows +v, lanes 16-31 -> rows +8+v; col = lane&15
    const int nc = col0 + nt * 16 + (lane & 15);
#pragma unroll
    for (int mt = 0; mt < MT; ++mt) {
        const int mbase = m0 + mt * 16 + (half ? 8 : 0);
#pragma unroll
        for (int v = 0; v < 8; ++v) {
            C[(long long)(mbase + v) * ldc + nc] = acc[mt][v];
        }
    }
}

// ---------------------------------------------------------------------------
// Edge gather/scale/scatter:  out[col[e], :] += dinv[row]*dinv[col] * xw[row, :]
// One wave per edge; b128 gathers, hardware f32 atomics into L2-resident out.
// ---------------------------------------------------------------------------
template <int C>
__global__ void edge_scatter(const float* __restrict__ xw,
                             float* __restrict__ out,
                             const long long* __restrict__ row,
                             const long long* __restrict__ col,
                             const float* __restrict__ dinv, int E) {
    const int wid  = (int)(((long long)blockIdx.x * blockDim.x + threadIdx.x) >> 5);
    const int lane = threadIdx.x & 31;
    if (wid >= E) return;

    const int r = (int)row[wid];
    const int c = (int)col[wid];
    const float nrm = dinv[r] * dinv[c];

    const float4* __restrict__ src = (const float4*)(xw + (long long)r * C);
    float* __restrict__ dst = out + (long long)c * C;
#pragma unroll
    for (int it = 0; it < (C >> 7); ++it) {
        const int j = it * 32 + lane;
        float4 v = src[j];
        unsafeAtomicAdd(dst + j * 4 + 0, v.x * nrm);
        unsafeAtomicAdd(dst + j * 4 + 1, v.y * nrm);
        unsafeAtomicAdd(dst + j * 4 + 2, v.z * nrm);
        unsafeAtomicAdd(dst + j * 4 + 3, v.w * nrm);
    }
}

// ---------------------------------------------------------------------------
// Epilogue 1: h = tanh(g1 + dinv^2 * xw1 + bias)   (self-loop folded in)
//   cols 0..127 -> b_f, cols 128..255 -> b_c.  In-place on g1. float4 wide.
// ---------------------------------------------------------------------------
__global__ void epilogue1(float4* __restrict__ g1,
                          const float4* __restrict__ xw1,
                          const float* __restrict__ dinv,
                          const float* __restrict__ bf,
                          const float* __restrict__ bc, int N) {
    long long idx = (long long)blockIdx.x * blockDim.x + threadIdx.x;
    if (idx >= (long long)N * 64) return;          // 64 float4 per row
    int i = (int)(idx >> 6);
    int j = (int)(idx & 63) * 4;                   // scalar column, mult of 4
    float di = dinv[i];
    float d2 = di * di;
    const float* bias = (j < 128) ? (bf + j) : (bc + (j - 128));
    float4 g = g1[idx], x = xw1[idx];
    g.x = tanhf(g.x + d2 * x.x + bias[0]);
    g.y = tanhf(g.y + d2 * x.y + bias[1]);
    g.z = tanhf(g.z + d2 * x.z + bias[2]);
    g.w = tanhf(g.w + d2 * x.w + bias[3]);
    g1[idx] = g;
}

// ---------------------------------------------------------------------------
// Epilogue 2: mean/logvar (+ self-loop + bias), reparameterize, write outputs.
//   d_out layout: [ z (N*64) | mean (N*64) | logvar (N*64) ]
// ---------------------------------------------------------------------------
__device__ __forceinline__ unsigned pcg_hash(unsigned x) {
    unsigned s = x * 747796405u + 2891336453u;
    unsigned w = ((s >> ((s >> 28u) + 4u)) ^ s) * 277803737u;
    return (w >> 22u) ^ w;
}

__global__ void epilogue2(const float* __restrict__ g2,
                          const float* __restrict__ hw,
                          const float* __restrict__ dinv,
                          const float* __restrict__ bm,
                          const float* __restrict__ blv,
                          float* __restrict__ out, int N) {
    long long idx = (long long)blockIdx.x * blockDim.x + threadIdx.x;
    const long long NN = (long long)N * 64;
    if (idx >= NN) return;
    int i = (int)(idx >> 6);
    int j = (int)(idx & 63);
    float di = dinv[i];
    float d2 = di * di;
    long long base = (long long)i * 128;
    float mean = g2[base + j]      + d2 * hw[base + j]      + bm[j];
    float lv   = g2[base + 64 + j] + d2 * hw[base + 64 + j] + blv[j];

    // hash-based Box-Muller standard normal
    unsigned h1 = pcg_hash((unsigned)idx * 2u + 1u);
    unsigned h2 = pcg_hash((unsigned)idx * 2u + 2u);
    float u1 = fmaxf((h1 >> 8) * (1.0f / 16777216.0f), 1e-7f);
    float u2 = (h2 >> 8) * (1.0f / 16777216.0f);
    float noise = sqrtf(-2.0f * logf(u1)) * cosf(6.28318530718f * u2);

    float z = noise * expf(0.5f * lv) + mean;
    out[idx]          = z;
    out[NN + idx]     = mean;
    out[2 * NN + idx] = lv;
}

// ---------------------------------------------------------------------------
// Host launch
// ---------------------------------------------------------------------------
extern "C" void kernel_launch(void* const* d_in, const int* in_sizes, int n_in,
                              void* d_out, int out_size, void* d_ws, size_t ws_size,
                              hipStream_t stream) {
    const float*     feature   = (const float*)d_in[0];     // [N,256]
    const float*     condition = (const float*)d_in[1];     // [N,128]
    const long long* ei        = (const long long*)d_in[2]; // [2,E]
    const float*     W_f  = (const float*)d_in[3];          // [256,128]
    const float*     b_f  = (const float*)d_in[4];          // [128]
    const float*     W_c  = (const float*)d_in[5];          // [128,128]
    const float*     b_c  = (const float*)d_in[6];          // [128]
    const float*     W_m  = (const float*)d_in[7];          // [256,64]
    const float*     b_m  = (const float*)d_in[8];          // [64]
    const float*     W_lv = (const float*)d_in[9];          // [256,64]
    const float*     b_lv = (const float*)d_in[10];         // [64]

    const int N = in_sizes[0] / 256;     // 50000
    const int E = in_sizes[2] / 2;       // 1600000
    const long long* row = ei;
    const long long* col = ei + E;

    // Workspace layout
    float*  ws    = (float*)d_ws;
    float*  deg   = ws;                                   // N floats (-> dinv)
    __bf16* Wpf   = (__bf16*)(deg + N);                   // 8*8*512  = 32768 bf16
    __bf16* Wpc   = Wpf + 32768;                          // 8*4*512  = 16384 bf16
    __bf16* Wpcat = Wpc + 16384;                          // 8*8*512  = 32768 bf16
    float*  xw1   = (float*)(Wpcat + 32768);              // N*256
    float*  g1    = xw1 + (long long)N * 256;             // N*256 (-> h)
    float*  hw    = xw1;                                  // N*128 (reuse xw1)
    float*  g2    = xw1 + (long long)N * 128;             // N*128 (reuse xw1)

    const int BT = 256;
    auto nb = [](long long n, int bt) { return (unsigned)((n + bt - 1) / bt); };

    // 1) degrees -> dinv  (self loop contributes 1.0)
    fill_f32x4<<<nb(N / 4, BT), BT, 0, stream>>>((float4*)deg, 1.0f, N / 4);
    deg_accum<<<nb(E, BT), BT, 0, stream>>>(col, deg, E);
    deg_rsqrt<<<nb(N, BT), BT, 0, stream>>>(deg, N);

    // 2) pre-pack weights into bf16 WMMA B-fragment layout
    pack_w_bf16<<<nb(32768, BT), BT, 0, stream>>>(W_f, W_f, 128, 256, 128, Wpf);
    pack_w_bf16<<<nb(16384, BT), BT, 0, stream>>>(W_c, W_c, 128, 128, 128, Wpc);
    pack_w_bf16<<<nb(32768, BT), BT, 0, stream>>>(W_m, W_lv, 64, 256, 128, Wpcat);

    // 3) layer-1 GEMMs: xw1[:, :128]=feature@W_f, xw1[:,128:]=condition@W_c
    {
        dim3 blk(32, 8);
        dim3 grd(N / 80, 1);   // 5 M-tiles (80 rows) per wave
        gemm_bf16_wmma<5><<<grd, blk, 0, stream>>>(feature,   256, Wpf, 128, xw1, 256, 0);
        gemm_bf16_wmma<5><<<grd, blk, 0, stream>>>(condition, 128, Wpc, 128, xw1, 256, 128);
    }

    // 4) fused 256-wide gather/scatter for both layer-1 convs
    fill_f32x4<<<nb((long long)N * 64, BT), BT, 0, stream>>>((float4*)g1, 0.0f, (long long)N * 64);
    edge_scatter<256><<<nb((long long)E * 32, BT), BT, 0, stream>>>(xw1, g1, row, col, deg, E);

    // 5) h = tanh(g1 + dinv^2*xw1 + bias)  (in place, g1 -> h)
    epilogue1<<<nb((long long)N * 64, BT), BT, 0, stream>>>((float4*)g1, (const float4*)xw1,
                                                            deg, b_f, b_c, N);

    // 6) hw = h @ [W_m|W_lv]   [N,128] (xw1 region reused)
    {
        dim3 blk(32, 8);
        dim3 grd(N / 80, 1);
        gemm_bf16_wmma<5><<<grd, blk, 0, stream>>>(g1, 256, Wpcat, 128, hw, 128, 0);
    }

    // 7) fused 128-wide gather/scatter for mean+logvar convs
    fill_f32x4<<<nb((long long)N * 32, BT), BT, 0, stream>>>((float4*)g2, 0.0f, (long long)N * 32);
    edge_scatter<128><<<nb((long long)E * 32, BT), BT, 0, stream>>>(hw, g2, row, col, deg, E);

    // 8) reparameterize + write (z, mean, logvar)
    epilogue2<<<nb((long long)N * 64, BT), BT, 0, stream>>>(g2, hw, deg, b_m, b_lv,
                                                            (float*)d_out, N);
}
